// StandaloneAlignment_22960895164516
// MI455X (gfx1250) — matrix operand order repaired and checked by reading, not compiled, not run on hardware
//
#include <hip/hip_runtime.h>
#include <math.h>

#define B_     8
#define T1_    2048
#define T2_    512
#define NMEL   100
#define NTEXT  512
#define C_     128
#define HID_   384
#define MAXPOS 1024

typedef __attribute__((ext_vector_type(16))) __bf16 v16bf;
typedef __attribute__((ext_vector_type(8)))  float  v8f;
typedef __attribute__((ext_vector_type(4)))  unsigned int v4u;
typedef __attribute__((ext_vector_type(4)))  int    v4i;
typedef __attribute__((ext_vector_type(8)))  int    v8i;

union BFrag { v16bf v; unsigned short u[16]; };

__device__ __forceinline__ unsigned short f2bf(float f) {
  unsigned int u = __float_as_uint(f);
  u += 0x7FFFu + ((u >> 16) & 1u);           // round-to-nearest-even
  return (unsigned short)(u >> 16);
}

__device__ __forceinline__ float wred_sum(float v) {
#pragma unroll
  for (int o = 16; o > 0; o >>= 1) v += __shfl_xor(v, o, 32);
  return v;
}
__device__ __forceinline__ float wred_max(float v) {
#pragma unroll
  for (int o = 16; o > 0; o >>= 1) v = fmaxf(v, __shfl_xor(v, o, 32));
  return v;
}

// ---------------- in-projection + LayerNorm (eps 1e-5) ----------------
// x: (B, IN, T) ; out: (B*T, C) token-major
__global__ void k_inproj(const float* __restrict__ x, const float* __restrict__ w,
                         const float* __restrict__ bias, const float* __restrict__ g,
                         const float* __restrict__ bb, float* __restrict__ out,
                         int IN, int T) {
  int t = blockIdx.x;
  int c = threadIdx.x;
  int b = t / T, tt = t - b * T;
  const float* xp = x + (size_t)b * IN * T + tt;
  float acc = bias[c];
  for (int i = 0; i < IN; ++i) acc += xp[(size_t)i * T] * w[i * C_ + c];
  __shared__ float red[C_];
  red[c] = acc; __syncthreads();
  for (int s = 64; s > 0; s >>= 1) { if (c < s) red[c] += red[c + s]; __syncthreads(); }
  float mean = red[0] * (1.f / C_);
  __syncthreads();
  float d = acc - mean;
  red[c] = d * d; __syncthreads();
  for (int s = 64; s > 0; s >>= 1) { if (c < s) red[c] += red[c + s]; __syncthreads(); }
  float var = red[0] * (1.f / C_);
  out[(size_t)t * C_ + c] = d * rsqrtf(var + 1e-5f) * g[c] + bb[c];
}

// ---------------- depthwise conv7 (pad 3) + LayerNorm (eps 1e-6) ----------------
__global__ void k_dwconv_ln(const float* __restrict__ xin, const float* __restrict__ dw,
                            const float* __restrict__ dwb, const float* __restrict__ g,
                            const float* __restrict__ bb, float* __restrict__ out, int T) {
  int t = blockIdx.x;
  int c = threadIdx.x;
  int b = t / T, tt = t - b * T;
  float acc = dwb[c];
#pragma unroll
  for (int j = 0; j < 7; ++j) {
    int s = tt + j - 3;
    if (s >= 0 && s < T) acc += xin[((size_t)(b * T + s)) * C_ + c] * dw[c * 7 + j];
  }
  __shared__ float red[C_];
  red[c] = acc; __syncthreads();
  for (int s = 64; s > 0; s >>= 1) { if (c < s) red[c] += red[c + s]; __syncthreads(); }
  float mean = red[0] * (1.f / C_);
  __syncthreads();
  float d = acc - mean;
  red[c] = d * d; __syncthreads();
  for (int s = 64; s > 0; s >>= 1) { if (c < s) red[c] += red[c + s]; __syncthreads(); }
  float var = red[0] * (1.f / C_);
  out[(size_t)t * C_ + c] = d * rsqrtf(var + 1e-6f) * g[c] + bb[c];
}

// ---------------- weight transpose: W (K,N) -> Wt (N,K) ----------------
__global__ void k_transpose(const float* __restrict__ w, float* __restrict__ wt,
                            int K, int N) {
  int id = blockIdx.x * blockDim.x + threadIdx.x;
  if (id < K * N) {
    int k = id / N, n = id - k * N;
    wt[(size_t)n * K + k] = w[id];
  }
}

// ---------------- bf16 WMMA GEMM: out = act(A' @ WtT + bias) [+ res] ----------------
// A: (ntok, K) f32; Wt: (N, K) f32 (pre-transposed); one wave per 16x16 tile.
// The 16xK weight tile is staged into LDS with the Tensor Data Mover (TDM)
// (1-row descriptor, data_size=4B, contiguous 16*K elements), then B-fragments
// are read from LDS. Optional per-(batch,channel) affine on A (GRN fold).
__global__ void k_gemm(const float* __restrict__ A, const float* __restrict__ Wt,
                       const float* __restrict__ bias, float* __restrict__ out,
                       const float* __restrict__ res, const float* __restrict__ ascale,
                       const float* __restrict__ ashift, int K, int N, int T, int actGelu) {
  __shared__ float btile[16 * HID_];   // up to 24KB (K<=384)
  int lane = threadIdx.x;
  int m = lane & 15, half = lane >> 4;
  int row = blockIdx.x * 16 + m;
  int bIdx = row / T;
  int ncol0 = blockIdx.y * 16;
  const float* wsrc = Wt + (size_t)ncol0 * K;
  const float* arow = A + (size_t)row * K;
  __builtin_prefetch(arow, 0, 3);

#if defined(__gfx1250__) && __has_builtin(__builtin_amdgcn_tensor_load_to_lds)
  // ---- TDM async copy: 16*K contiguous f32 from global -> LDS ----
  unsigned nelem = (unsigned)(16 * K);                 // <= 6144, fits 16 bits
  unsigned ldsByte = (unsigned)(size_t)(void*)btile;
  unsigned long long ga = (unsigned long long)(size_t)(const void*)wsrc;
  v4u g0;
  g0.x = 1u;                                           // count=1 (valid descriptor)
  g0.y = ldsByte;                                      // lds_addr
  g0.z = (unsigned)(ga & 0xFFFFFFFFull);               // global_addr[31:0]
  g0.w = (unsigned)((ga >> 32) & 0x01FFFFFFull) | 0x80000000u;  // addr[56:32] | type=2
  v8i g1;
  g1[0] = (int)(2u << 16);                             // data_size=2 (4 bytes)
  g1[1] = (int)((nelem & 0xFFFFu) << 16);              // tensor_dim0[15:0]
  g1[2] = (int)((nelem >> 16) & 0xFFFFu) | (int)(1u << 16);  // dim0 hi | tensor_dim1=1
  g1[3] = (int)((nelem & 0xFFFFu) << 16);              // tile_dim0 = 16*K
  g1[4] = 1;                                           // tile_dim1 = 1
  g1[5] = (int)nelem;                                  // tensor_dim0_stride lo32
  g1[6] = (int)((nelem & 0xFFFFu) << 16);              // stride hi | dim1_stride lo16
  g1[7] = (int)(nelem >> 16);                          // dim1_stride hi
  v4i z4 = {0, 0, 0, 0};
#if __clang_major__ >= 23
  v8i z8 = {0, 0, 0, 0, 0, 0, 0, 0};
  __builtin_amdgcn_tensor_load_to_lds(g0, g1, z4, z4, z8, 0);
#else
  __builtin_amdgcn_tensor_load_to_lds(g0, g1, z4, z4, 0);
#endif
  __builtin_amdgcn_s_wait_tensorcnt(0);
#else
  for (int idx = lane; idx < 16 * K; idx += 32) btile[idx] = wsrc[idx];
  __syncthreads();
#endif

  v8f acc = {0.f, 0.f, 0.f, 0.f, 0.f, 0.f, 0.f, 0.f};
  const float* scp = ascale ? (ascale + (size_t)bIdx * K) : (const float*)0;
  const float* brow = btile + (size_t)m * K;           // weight row n=m from LDS
  for (int k0 = 0; k0 < K; k0 += 32) {
    BFrag af, bf;
#pragma unroll
    for (int v = 0; v < 8; ++v) {
      int ka = k0 + ((v >> 2) << 4) + (half << 3) + ((v & 3) << 1);   // A frag K map
      float2 a2 = *(const float2*)(arow + ka);
      if (scp) {
        a2.x = a2.x * scp[ka]     + ashift[ka];
        a2.y = a2.y * scp[ka + 1] + ashift[ka + 1];
      }
      af.u[2 * v]     = f2bf(a2.x);
      af.u[2 * v + 1] = f2bf(a2.y);
      int kb = k0 + ((v >> 2) << 4) + (half << 3) + ((v & 3) << 1);   // B frag K map
      float2 b2 = *(const float2*)(brow + kb);                        // ds_load_b64
      bf.u[2 * v]     = f2bf(b2.x);
      bf.u[2 * v + 1] = f2bf(b2.y);
    }
    acc = __builtin_amdgcn_wmma_f32_16x16x32_bf16(false, af.v, false, bf.v,
                                                  (short)0, acc, false, false);
  }
  int ncg = ncol0 + m;
#pragma unroll
  for (int r = 0; r < 8; ++r) {
    int tok = blockIdx.x * 16 + r + half * 8;
    float vv = acc[r] + bias[ncg];
    if (actGelu) vv = 0.5f * vv * (1.f + erff(vv * 0.70710678118f));  // exact GELU
    size_t oi = (size_t)tok * N + ncg;
    if (res) vv += res[oi];
    out[oi] = vv;
  }
}

// ---------------- GRN stage 1: gx[b,c] = sum_t h^2 ----------------
__global__ void k_colsumsq(const float* __restrict__ h, float* __restrict__ gx, int T) {
  int b = blockIdx.x;
  int c = blockIdx.y * 128 + threadIdx.x;
  const float* hp = h + (size_t)b * T * HID_ + c;
  float s = 0.f;
  for (int t = 0; t < T; ++t) { float v = hp[(size_t)t * HID_]; s += v * v; }
  gx[b * HID_ + c] = s;
}

// ---------------- GRN stage 2: scale[b,c] = 1 + g[c]*(sqrt(gx)/(mean_c sqrt(gx)+1e-6)) ----------------
__global__ void k_grn_scale(const float* __restrict__ gx, const float* __restrict__ gg,
                            float* __restrict__ scaleB) {
  int b = blockIdx.x, tid = threadIdx.x;
  float part = 0.f;
#pragma unroll
  for (int i = 0; i < 3; ++i) part += sqrtf(gx[b * HID_ + tid + i * 128]);
  __shared__ float red[128];
  red[tid] = part; __syncthreads();
  for (int s = 64; s > 0; s >>= 1) { if (tid < s) red[tid] += red[tid + s]; __syncthreads(); }
  float mean = red[0] * (1.f / HID_);
#pragma unroll
  for (int i = 0; i < 3; ++i) {
    int c = tid + i * 128;
    float gv = sqrtf(gx[b * HID_ + c]);
    scaleB[b * HID_ + c] = 1.f + gg[c] * (gv / (mean + 1e-6f));
  }
}

// ---------------- RMSNorm + scaled squared-norm (for cdist) ----------------
__global__ void k_rmsnorm(const float* __restrict__ xin, const float* __restrict__ g,
                          float* __restrict__ enc, float* __restrict__ nrm) {
  int t = blockIdx.x, c = threadIdx.x;
  float x = xin[(size_t)t * C_ + c];
  __shared__ float red[C_];
  red[c] = x * x; __syncthreads();
  for (int s = 64; s > 0; s >>= 1) { if (c < s) red[c] += red[c + s]; __syncthreads(); }
  float rms = rsqrtf(red[0] * (1.f / C_) + 1e-6f);
  __syncthreads();
  float y = x * rms * g[c];
  enc[(size_t)t * C_ + c] = y;
  red[c] = y * y; __syncthreads();
  for (int s = 64; s > 0; s >>= 1) { if (c < s) red[c] += red[c + s]; __syncthreads(); }
  if (c == 0) nrm[t] = red[0] * (1.f / C_);   // |y * C^-0.5|^2
}

// ---------------- RoPE ----------------
__global__ void k_rope(const float* __restrict__ enc, float* __restrict__ rot,
                       const float* __restrict__ tes, int T, int isKey) {
  int t = blockIdx.x;
  int i = threadIdx.x;              // 0..63 frequency pair
  int b = t / T, tt = t - b * T;
  int pos;
  if (isKey) {
    int pi = (int)((float)tt * tes[b]);   // truncation like astype(int32)
    pos = min(max(pi, 0), MAXPOS - 1);
  } else {
    pos = min(tt, MAXPOS - 1);
  }
  float freq = __expf(-(float)i * (9.210340371976184f / 64.f));  // 10000^(-2i/128)
  float ang = (float)pos * freq;
  float cs = __cosf(ang), sn = __sinf(ang);
  float x1 = enc[(size_t)t * C_ + 2 * i];
  float x2 = enc[(size_t)t * C_ + 2 * i + 1];
  rot[(size_t)t * C_ + 2 * i]     = x1 * cs - x2 * sn;
  rot[(size_t)t * C_ + 2 * i + 1] = x1 * sn + x2 * cs;
}

// ---------------- attention: dual cdist via WMMA + softmax + prior renorm ----------------
__global__ void __launch_bounds__(256) k_attn(
    const float* __restrict__ qe, const float* __restrict__ qr, const float* __restrict__ qn,
    const float* __restrict__ ke, const float* __restrict__ kr, const float* __restrict__ kn,
    const float* __restrict__ mask, const float* __restrict__ prior, float* __restrict__ out) {
  __shared__ float sc[16 * T2_];   // 32KB score slab (16 q-rows x 512 keys)
  int tid = threadIdx.x;
  int wave = tid >> 5, lane = tid & 31;
  int m = lane & 15, half = lane >> 4;
  int b = blockIdx.y;
  int t1b = blockIdx.x * 16;
  const float* qrow  = qe + ((size_t)(b * T1_ + t1b + m)) * C_;
  const float* qrrow = qr + ((size_t)(b * T1_ + t1b + m)) * C_;
  __builtin_prefetch(qrow, 0, 3);
  __builtin_prefetch(qrrow, 0, 3);
  for (int it = 0; it < 4; ++it) {
    int nt = wave * 4 + it;        // key tile 0..31
    const float* krow  = ke + ((size_t)(b * T2_ + nt * 16 + m)) * C_;
    const float* krrow = kr + ((size_t)(b * T2_ + nt * 16 + m)) * C_;
    v8f acc  = {0.f, 0.f, 0.f, 0.f, 0.f, 0.f, 0.f, 0.f};
    v8f accr = {0.f, 0.f, 0.f, 0.f, 0.f, 0.f, 0.f, 0.f};
    for (int k0 = 0; k0 < C_; k0 += 32) {
      BFrag af, afr, bf, bfr;
#pragma unroll
      for (int v = 0; v < 8; ++v) {
        int ka = k0 + ((v >> 2) << 4) + (half << 3) + ((v & 3) << 1);
        float2 a2  = *(const float2*)(qrow + ka);
        float2 a2r = *(const float2*)(qrrow + ka);
        af.u[2 * v]      = f2bf(a2.x);  af.u[2 * v + 1]  = f2bf(a2.y);
        afr.u[2 * v]     = f2bf(a2r.x); afr.u[2 * v + 1] = f2bf(a2r.y);
        int kb = k0 + ((v >> 2) << 4) + (half << 3) + ((v & 3) << 1);
        float2 b2  = *(const float2*)(krow + kb);
        float2 b2r = *(const float2*)(krrow + kb);
        bf.u[2 * v]      = f2bf(b2.x);  bf.u[2 * v + 1]  = f2bf(b2.y);
        bfr.u[2 * v]     = f2bf(b2r.x); bfr.u[2 * v + 1] = f2bf(b2r.y);
      }
      acc  = __builtin_amdgcn_wmma_f32_16x16x32_bf16(false, af.v,  false, bf.v,
                                                     (short)0, acc,  false, false);
      accr = __builtin_amdgcn_wmma_f32_16x16x32_bf16(false, afr.v, false, bfr.v,
                                                     (short)0, accr, false, false);
    }
#pragma unroll
    for (int r = 0; r < 8; ++r) {
      int rloc = r + half * 8;
      int t1 = t1b + rloc;
      int t2 = nt * 16 + m;
      float qq = qn[b * T1_ + t1];
      float kk = kn[b * T2_ + t2];
      // |q_rot| == |q_enc| exactly (RoPE is a rotation) -> reuse norms
      float d  = sqrtf(fmaxf(qq + kk - 2.f * acc[r]  * (1.f / C_), 1e-12f));
      float dr = sqrtf(fmaxf(qq + kk - 2.f * accr[r] * (1.f / C_), 1e-12f));
      sc[rloc * T2_ + t2] = -(0.9f * d + 0.1f * dr) + mask[b * T2_ + t2];
    }
  }
  __syncthreads();
#pragma unroll
  for (int rr = 0; rr < 2; ++rr) {
    int rloc = wave + rr * 8;
    int t1 = t1b + rloc;
    const float* pr = prior + ((size_t)b * T1_ + t1) * T2_;
    __builtin_prefetch(pr, 0, 3);
    float mx = -1e30f;
    for (int j = lane; j < T2_; j += 32) mx = fmaxf(mx, sc[rloc * T2_ + j]);
    mx = wred_max(mx);
    float sm = 0.f;
    for (int j = lane; j < T2_; j += 32) sm += __expf(sc[rloc * T2_ + j] - mx);
    sm = wred_sum(sm);
    float inv_sm = 1.f / sm;
    float s2 = 0.f;
    for (int j = lane; j < T2_; j += 32) {
      float a = __expf(sc[rloc * T2_ + j] - mx) * inv_sm;
      a = fmaxf(a, 1e-8f) * fmaxf(pr[j], 1e-8f);   // exp(log(max)+log(max))
      sc[rloc * T2_ + j] = a;
      s2 += a;
    }
    s2 = wred_sum(s2);
    float inv = 1.f / (s2 + 1e-8f);
    float* op = out + ((size_t)b * T1_ + t1) * T2_;
    for (int j = lane; j < T2_; j += 32) op[j] = sc[rloc * T2_ + j] * inv;
  }
}

extern "C" void kernel_launch(void* const* d_in, const int* in_sizes, int n_in,
                              void* d_out, int out_size, void* d_ws, size_t ws_size,
                              hipStream_t stream) {
  (void)in_sizes; (void)n_in; (void)out_size; (void)ws_size;
  auto F = [&](int i) { return (const float*)d_in[i]; };
  const float* queries = F(0);
  const float* keys    = F(1);
  const float* tes     = F(2);
  const float* mask    = F(3);
  const float* prior   = F(4);

  struct Blk { const float *dw_w, *dw_b, *ln_g, *ln_b, *w1, *b1, *grn_g, *grn_b, *w2, *b2; };
  struct Branch {
    const float *in_w, *in_b, *in_ln_g, *in_ln_b;
    Blk blk[3];
    const float *out_w, *out_b, *rms_g;
  };
  auto loadBranch = [&](int i) {
    Branch p;
    p.in_w = F(i++); p.in_b = F(i++); p.in_ln_g = F(i++); p.in_ln_b = F(i++);
    for (int l = 0; l < 3; ++l) {
      p.blk[l].dw_w = F(i++); p.blk[l].dw_b = F(i++);
      p.blk[l].ln_g = F(i++); p.blk[l].ln_b = F(i++);
      p.blk[l].w1 = F(i++);   p.blk[l].b1 = F(i++);
      p.blk[l].grn_g = F(i++); p.blk[l].grn_b = F(i++);
      p.blk[l].w2 = F(i++);   p.blk[l].b2 = F(i++);
    }
    p.out_w = F(i++); p.out_b = F(i++); p.rms_g = F(i++);
    return p;
  };
  Branch qp = loadBranch(5);
  Branch kp = loadBranch(42);

  float* W = (float*)d_ws;
  size_t o = 0;
  auto alloc = [&](size_t n) { float* p = W + o; o += n; return p; };
  float* bufX = alloc((size_t)B_ * T1_ * C_);
  float* bufY = alloc((size_t)B_ * T1_ * C_);
  float* bufH = alloc((size_t)B_ * T1_ * HID_);
  float* wT   = alloc((size_t)HID_ * C_);
  float* gx   = alloc((size_t)B_ * HID_);
  float* scB  = alloc((size_t)B_ * HID_);
  float* qenc = alloc((size_t)B_ * T1_ * C_);
  float* qrot = alloc((size_t)B_ * T1_ * C_);
  float* qn   = alloc((size_t)B_ * T1_);
  float* kenc = alloc((size_t)B_ * T2_ * C_);
  float* krot = alloc((size_t)B_ * T2_ * C_);
  float* kn   = alloc((size_t)B_ * T2_);

  auto encode = [&](const Branch& p, const float* xin, int IN, int T,
                    float* enc, float* rot, float* nrm, int isKey) {
    int ntok = B_ * T;
    k_inproj<<<ntok, 128, 0, stream>>>(xin, p.in_w, p.in_b, p.in_ln_g, p.in_ln_b, bufX, IN, T);
    for (int l = 0; l < 3; ++l) {
      const Blk& bk = p.blk[l];
      k_dwconv_ln<<<ntok, 128, 0, stream>>>(bufX, bk.dw_w, bk.dw_b, bk.ln_g, bk.ln_b, bufY, T);
      k_transpose<<<(C_ * HID_ + 255) / 256, 256, 0, stream>>>(bk.w1, wT, C_, HID_);
      k_gemm<<<dim3(ntok / 16, HID_ / 16), 32, 0, stream>>>(
          bufY, wT, bk.b1, bufH, (const float*)0, (const float*)0, (const float*)0,
          C_, HID_, T, 1);
      k_colsumsq<<<dim3(B_, HID_ / 128), 128, 0, stream>>>(bufH, gx, T);
      k_grn_scale<<<B_, 128, 0, stream>>>(gx, bk.grn_g, scB);
      k_transpose<<<(HID_ * C_ + 255) / 256, 256, 0, stream>>>(bk.w2, wT, HID_, C_);
      k_gemm<<<dim3(ntok / 16, C_ / 16), 32, 0, stream>>>(
          bufH, wT, bk.b2, bufX, bufX, scB, bk.grn_b, HID_, C_, T, 0);
    }
    k_transpose<<<(C_ * C_ + 255) / 256, 256, 0, stream>>>(p.out_w, wT, C_, C_);
    k_gemm<<<dim3(ntok / 16, C_ / 16), 32, 0, stream>>>(
        bufX, wT, p.out_b, bufY, (const float*)0, (const float*)0, (const float*)0,
        C_, C_, T, 0);
    k_rmsnorm<<<ntok, 128, 0, stream>>>(bufY, p.rms_g, enc, nrm);
    k_rope<<<ntok, 64, 0, stream>>>(enc, rot, tes, T, isKey);
  };

  encode(kp, keys,    NTEXT, T2_, kenc, krot, kn, 1);
  encode(qp, queries, NMEL,  T1_, qenc, qrot, qn, 0);

  k_attn<<<dim3(T1_ / 16, B_), 256, 0, stream>>>(qenc, qrot, qn, kenc, krot, kn,
                                                 mask, prior, (float*)d_out);
}
